// DocREModel_5076651344288
// MI455X (gfx1250) — matrix operand order repaired and checked by reading, not compiled, not run on hardware
//
#include <hip/hip_runtime.h>
#include <hip/hip_bf16.h>

// ---- problem constants ----
#define B_   2
#define L_   1024
#define H_   768
#define A_   12
#define E_   24
#define MM_  8      // mentions per entity
#define P_   552
#define NL_  97
#define NLP_ 128    // NL padded for GEMM B width
#define NP_  (B_*P_)   // 1104 rows
#define EMBD 768
#define BLK_ 64
#define KB_  (EMBD*BLK_)  // 49152

typedef __attribute__((ext_vector_type(16))) __bf16 bf16x16;
typedef __attribute__((ext_vector_type(8)))  float  f32x8;
typedef __attribute__((ext_vector_type(4)))  unsigned int u32x4;

union Frag16 { bf16x16 v; u32x4 q[2]; };
union Acc8   { f32x8 v; float f[8]; };
union Pack8  { u32x4 q; __bf16 h[8]; };

// =====================================================================
// bf16 WMMA GEMM:  C[M,Nout] = act( A[M,K] @ B[K,N] + bias )
// A row-major bf16 (M,K); B supplied PRE-TRANSPOSED as Bt (N,K) bf16.
// Block tile 128x128, BK=32, 256 threads = 8 waves; wave tile 32x64
// (2 m-tiles x 4 n-tiles of 16x16). Double-buffered LDS, one barrier
// per K-step. Fragment layouts per CDNA5 ISA 7.12.2.
// =====================================================================
__global__ __launch_bounds__(256)
void gemm_bf16_wmma(const __bf16* __restrict__ Ag, const __bf16* __restrict__ Bt,
                    const float* __restrict__ bias,
                    __bf16* __restrict__ Cb, float* __restrict__ Cf,
                    int M, int N, int K, int Nout, int ldc, int act)
{
    __shared__ __align__(16) unsigned short sA[2][128 * 32];   // [m][k]
    __shared__ __align__(16) unsigned short sB[2][128 * 32];   // [n][k]

    const int tid  = threadIdx.x;
    const int lane = tid & 31;
    const int wave = tid >> 5;
    const int wm   = wave & 3;        // 0..3  (32-row slab)
    const int wn   = wave >> 2;       // 0..1  (64-col slab)
    const int hlf  = lane >> 4;       // lane half (0/1)
    const int l15  = lane & 15;

    const int m0 = blockIdx.y * 128;
    const int n0 = blockIdx.x * 128;

    Acc8 acc[2][4];
#pragma unroll
    for (int i = 0; i < 2; ++i)
#pragma unroll
        for (int j = 0; j < 4; ++j)
#pragma unroll
            for (int r = 0; r < 8; ++r) acc[i][j].f[r] = 0.f;

    // Tile staging map (identical for A and Bt): thread t -> row t>>1 (0..127),
    // 16 bf16 at col (t&1)*16. Two b128 global loads, two b128 LDS stores.
    const int row  = tid >> 1;
    const int col  = (tid & 1) * 16;
    int a_grow = m0 + row; if (a_grow > M - 1) a_grow = M - 1;   // M-tail clamp
    const __bf16* aPtr = Ag + (size_t)a_grow * K + col;
    const __bf16* bPtr = Bt + (size_t)(n0 + row) * K + col;
    const int sOff = row * 32 + col;

    // ---- preload tile 0 ----
    u32x4 ra0 = *reinterpret_cast<const u32x4*>(aPtr);
    u32x4 ra1 = *reinterpret_cast<const u32x4*>(aPtr + 8);
    u32x4 rb0 = *reinterpret_cast<const u32x4*>(bPtr);
    u32x4 rb1 = *reinterpret_cast<const u32x4*>(bPtr + 8);
    *reinterpret_cast<u32x4*>(&sA[0][sOff])     = ra0;
    *reinterpret_cast<u32x4*>(&sA[0][sOff + 8]) = ra1;
    *reinterpret_cast<u32x4*>(&sB[0][sOff])     = rb0;
    *reinterpret_cast<u32x4*>(&sB[0][sOff + 8]) = rb1;
    __syncthreads();

    int cur = 0;
    for (int k0 = 0; k0 < K; k0 += 32) {
        const bool nxt = (k0 + 32) < K;
        if (nxt) {
            ra0 = *reinterpret_cast<const u32x4*>(aPtr + k0 + 32);
            ra1 = *reinterpret_cast<const u32x4*>(aPtr + k0 + 40);
            rb0 = *reinterpret_cast<const u32x4*>(bPtr + k0 + 32);
            rb1 = *reinterpret_cast<const u32x4*>(bPtr + k0 + 40);
            if (k0 + 64 < K) {
                __builtin_prefetch(aPtr + k0 + 64, 0, 3);
                __builtin_prefetch(bPtr + k0 + 64, 0, 3);
            }
        }

        // ---- fragments (exact per-lane ISA layout, b128 LDS reads) ----
        Frag16 af[2], bfr[4];
#pragma unroll
        for (int i = 0; i < 2; ++i) {
            const int mr = wm * 32 + i * 16 + l15;
            af[i].q[0] = *reinterpret_cast<const u32x4*>(&sA[cur][mr * 32 + hlf * 8]);
            af[i].q[1] = *reinterpret_cast<const u32x4*>(&sA[cur][mr * 32 + 16 + hlf * 8]);
        }
#pragma unroll
        for (int j = 0; j < 4; ++j) {
            const int nc = wn * 64 + j * 16 + l15;
            bfr[j].q[0] = *reinterpret_cast<const u32x4*>(&sB[cur][nc * 32 + hlf * 8]);
            bfr[j].q[1] = *reinterpret_cast<const u32x4*>(&sB[cur][nc * 32 + 16 + hlf * 8]);
        }

#pragma unroll
        for (int i = 0; i < 2; ++i)
#pragma unroll
            for (int j = 0; j < 4; ++j)
                acc[i][j].v = __builtin_amdgcn_wmma_f32_16x16x32_bf16(
                    false, af[i].v, false, bfr[j].v,
                    (short)0, acc[i][j].v, false, false);

        if (nxt) {
            *reinterpret_cast<u32x4*>(&sA[cur ^ 1][sOff])     = ra0;
            *reinterpret_cast<u32x4*>(&sA[cur ^ 1][sOff + 8]) = ra1;
            *reinterpret_cast<u32x4*>(&sB[cur ^ 1][sOff])     = rb0;
            *reinterpret_cast<u32x4*>(&sB[cur ^ 1][sOff + 8]) = rb1;
        }
        __syncthreads();
        cur ^= 1;
    }

#pragma unroll
    for (int i = 0; i < 2; ++i) {
#pragma unroll
        for (int j = 0; j < 4; ++j) {
            const int ncol = n0 + wn * 64 + j * 16 + l15;
            if (ncol >= Nout) continue;
            const float bv = bias ? bias[ncol] : 0.f;
#pragma unroll
            for (int r = 0; r < 8; ++r) {
                const int mrow = m0 + wm * 32 + i * 16 + r + hlf * 8;
                if (mrow >= M) continue;
                float v = acc[i][j].f[r] + bv;
                if (act == 1) v = tanhf(v);
                if (Cf) Cf[(size_t)mrow * ldc + ncol] = v;
                if (Cb) Cb[(size_t)mrow * ldc + ncol] = (__bf16)v;
            }
        }
    }
}

// =====================================================================
// FUSED block-bilinear GEMM:
//   C[M,Nout] = ( X @ B ) + bias,   X[n, k*4096+bi*64+ci] = hv[n,k*64+bi]*tv[n,k*64+ci]
// X is never materialized: each staged A-tile row within a BK=32 chunk
// (fixed k,bi) is one hv scalar times 32 contiguous tv values.
// hv/tv are (M,768) bf16 (tiny, cache-resident) -> no HBM streaming of X.
// Same WMMA pipeline as gemm_bf16_wmma; K fixed = 49152.
// =====================================================================
__global__ __launch_bounds__(256)
void gemm_bilinear_wmma(const __bf16* __restrict__ hv, const __bf16* __restrict__ tv,
                        const __bf16* __restrict__ Bt, const float* __restrict__ bias,
                        __bf16* __restrict__ Cb, float* __restrict__ Cf,
                        int M, int Nout, int ldc)
{
    const int K = KB_;
    __shared__ __align__(16) unsigned short sA[2][128 * 32];
    __shared__ __align__(16) unsigned short sB[2][128 * 32];

    const int tid  = threadIdx.x;
    const int lane = tid & 31;
    const int wave = tid >> 5;
    const int wm   = wave & 3;
    const int wn   = wave >> 2;
    const int hlf  = lane >> 4;
    const int l15  = lane & 15;

    const int m0 = blockIdx.y * 128;
    const int n0 = blockIdx.x * 128;

    Acc8 acc[2][4];
#pragma unroll
    for (int i = 0; i < 2; ++i)
#pragma unroll
        for (int j = 0; j < 4; ++j)
#pragma unroll
            for (int r = 0; r < 8; ++r) acc[i][j].f[r] = 0.f;

    const int row  = tid >> 1;
    const int col  = (tid & 1) * 16;
    int a_grow = m0 + row; if (a_grow > M - 1) a_grow = M - 1;
    const __bf16* hRow = hv + (size_t)a_grow * EMBD;
    const __bf16* tRow = tv + (size_t)a_grow * EMBD;
    const __bf16* bPtr = Bt + (size_t)(n0 + row) * K + col;
    const int sOff = row * 32 + col;

    // generate 16 bf16 of the X tile for chunk k0 handled by this thread
    auto stageA = [&](int k0, u32x4& o0, u32x4& o1) {
        const int k   = k0 >> 12;             // EMB block (0..11)
        const int bi  = (k0 >> 6) & 63;       // hv element in block
        const int ci0 = (k0 & 63) + col;      // tv start (0/16/32/48)
        const float hs = (float)hRow[k * 64 + bi];
        Pack8 t0, t1, r0, r1;
        t0.q = *reinterpret_cast<const u32x4*>(tRow + k * 64 + ci0);
        t1.q = *reinterpret_cast<const u32x4*>(tRow + k * 64 + ci0 + 8);
#pragma unroll
        for (int j = 0; j < 8; ++j) {
            r0.h[j] = (__bf16)(hs * (float)t0.h[j]);
            r1.h[j] = (__bf16)(hs * (float)t1.h[j]);
        }
        o0 = r0.q; o1 = r1.q;
    };

    // ---- preload tile 0 ----
    u32x4 ra0, ra1;
    stageA(0, ra0, ra1);
    u32x4 rb0 = *reinterpret_cast<const u32x4*>(bPtr);
    u32x4 rb1 = *reinterpret_cast<const u32x4*>(bPtr + 8);
    *reinterpret_cast<u32x4*>(&sA[0][sOff])     = ra0;
    *reinterpret_cast<u32x4*>(&sA[0][sOff + 8]) = ra1;
    *reinterpret_cast<u32x4*>(&sB[0][sOff])     = rb0;
    *reinterpret_cast<u32x4*>(&sB[0][sOff + 8]) = rb1;
    __syncthreads();

    int cur = 0;
    for (int k0 = 0; k0 < K; k0 += 32) {
        const bool nxt = (k0 + 32) < K;
        if (nxt) {
            stageA(k0 + 32, ra0, ra1);
            rb0 = *reinterpret_cast<const u32x4*>(bPtr + k0 + 32);
            rb1 = *reinterpret_cast<const u32x4*>(bPtr + k0 + 40);
            if (k0 + 64 < K) __builtin_prefetch(bPtr + k0 + 64, 0, 3);
        }

        Frag16 af[2], bfr[4];
#pragma unroll
        for (int i = 0; i < 2; ++i) {
            const int mr = wm * 32 + i * 16 + l15;
            af[i].q[0] = *reinterpret_cast<const u32x4*>(&sA[cur][mr * 32 + hlf * 8]);
            af[i].q[1] = *reinterpret_cast<const u32x4*>(&sA[cur][mr * 32 + 16 + hlf * 8]);
        }
#pragma unroll
        for (int j = 0; j < 4; ++j) {
            const int nc = wn * 64 + j * 16 + l15;
            bfr[j].q[0] = *reinterpret_cast<const u32x4*>(&sB[cur][nc * 32 + hlf * 8]);
            bfr[j].q[1] = *reinterpret_cast<const u32x4*>(&sB[cur][nc * 32 + 16 + hlf * 8])
        }

#pragma unroll
        for (int i = 0; i < 2; ++i)
#pragma unroll
            for (int j = 0; j < 4; ++j)
                acc[i][j].v = __builtin_amdgcn_wmma_f32_16x16x32_bf16(
                    false, af[i].v, false, bfr[j].v,
                    (short)0, acc[i][j].v, false, false);

        if (nxt) {
            *reinterpret_cast<u32x4*>(&sA[cur ^ 1][sOff])     = ra0;
            *reinterpret_cast<u32x4*>(&sA[cur ^ 1][sOff + 8]) = ra1;
            *reinterpret_cast<u32x4*>(&sB[cur ^ 1][sOff])     = rb0;
            *reinterpret_cast<u32x4*>(&sB[cur ^ 1][sOff + 8]) = rb1;
        }
        __syncthreads();
        cur ^= 1;
    }

#pragma unroll
    for (int i = 0; i < 2; ++i) {
#pragma unroll
        for (int j = 0; j < 4; ++j) {
            const int ncol = n0 + wn * 64 + j * 16 + l15;
            if (ncol >= Nout) continue;
            const float bv = bias ? bias[ncol] : 0.f;
#pragma unroll
            for (int r = 0; r < 8; ++r) {
                const int mrow = m0 + wm * 32 + i * 16 + r + hlf * 8;
                if (mrow >= M) continue;
                const float v = acc[i][j].f[r] + bv;
                if (Cf) Cf[(size_t)mrow * ldc + ncol] = v;
                if (Cb) Cb[(size_t)mrow * ldc + ncol] = (__bf16)v;
            }
        }
    }
}

// =====================================================================
// f32 (K,N) -> bf16 (Nout,K) transpose-convert (LDS 32x32 tiles).
// Rows of the output beyond N are zero-padded. blockIdx.z batches.
// =====================================================================
__global__ __launch_bounds__(256)
void cvt_transpose_bf16(const float* __restrict__ in, __bf16* __restrict__ out,
                        int K, int N, int Nout)
{
    __shared__ float tile[32][33];
    const int k0 = blockIdx.y * 32;
    const int n0 = blockIdx.x * 32;
    in  += (size_t)blockIdx.z * K * N;
    out += (size_t)blockIdx.z * Nout * K;
    const int tx = threadIdx.x & 31;
    const int ty = threadIdx.x >> 5;          // 0..7
#pragma unroll
    for (int i = 0; i < 32; i += 8) {
        const int n = n0 + tx;
        tile[ty + i][tx] = (n < N) ? in[(size_t)(k0 + ty + i) * N + n] : 0.f;
    }
    __syncthreads();
#pragma unroll
    for (int i = 0; i < 32; i += 8)
        out[(size_t)(n0 + ty + i) * K + k0 + tx] = (__bf16)tile[tx][ty + i];
}

// =====================================================================
// Small kernels
// =====================================================================
__global__ void ent_emb_kernel(const float* __restrict__ seq,
                               const float* __restrict__ mask,
                               const int* __restrict__ midx,
                               float* __restrict__ ent_emb)
{
    const int be = blockIdx.x;          // b*E + e
    const int b  = be / E_;
    __shared__ int   pos[MM_];
    __shared__ float msk[MM_];
    if (threadIdx.x < MM_) {
        pos[threadIdx.x] = midx[be * MM_ + threadIdx.x] + 1;   // OFFSET
        msk[threadIdx.x] = mask[be * MM_ + threadIdx.x];
    }
    __syncthreads();
    for (int h = threadIdx.x; h < H_; h += blockDim.x) {
        float x[MM_];
        float mx = -3.0e38f;
#pragma unroll
        for (int m = 0; m < MM_; ++m) {
            const float v = seq[((size_t)(b * L_ + pos[m])) * H_ + h];
            x[m] = (msk[m] > 0.f) ? v : -1e30f;
            mx = fmaxf(mx, x[m]);
        }
        float s = 0.f;
#pragma unroll
        for (int m = 0; m < MM_; ++m) s += __expf(x[m] - mx);
        ent_emb[(size_t)be * H_ + h] = mx + __logf(s);
    }
}

__global__ void ent_att_kernel(const float* __restrict__ att,
                               const float* __restrict__ mask,
                               const int* __restrict__ midx,
                               float* __restrict__ ent_att)
{
    const int bea = blockIdx.x;                 // (b*E+e)*A + a
    const int a  = bea % A_;
    const int be = bea / A_;
    const int b  = be / E_;
    __shared__ int   pos[MM_];
    __shared__ float msk[MM_];
    __shared__ float inv;
    if (threadIdx.x < MM_) {
        pos[threadIdx.x] = midx[be * MM_ + threadIdx.x] + 1;
        msk[threadIdx.x] = mask[be * MM_ + threadIdx.x];
    }
    __syncthreads();
    if (threadIdx.x == 0) {
        float c = 0.f;
        for (int m = 0; m < MM_; ++m) c += msk[m];
        inv = 1.f / fmaxf(c, 1.f);
    }
    __syncthreads();
    const float* base = att + ((size_t)(b * A_ + a)) * L_ * L_;
    for (int l = threadIdx.x; l < L_; l += blockDim.x) {
        float s = 0.f;
#pragma unroll
        for (int m = 0; m < MM_; ++m) s += base[(size_t)pos[m] * L_ + l] * msk[m];
        ent_att[(size_t)bea * L_ + l] = s * inv;
    }
}

__global__ __launch_bounds__(256)
void ht_att_kernel(const float* __restrict__ ent_att,
                   const int* __restrict__ hts,
                   __bf16* __restrict__ ht_bf)
{
    const int bp = blockIdx.x;     // b*P + p
    const int b  = bp / P_;
    const int hI = hts[bp * 2 + 0];
    const int tI = hts[bp * 2 + 1];
    const float* ha = ent_att + ((size_t)(b * E_ + hI)) * A_ * L_;
    const float* ta = ent_att + ((size_t)(b * E_ + tI)) * A_ * L_;

    float loc[4];
    float lsum = 0.f;
#pragma unroll
    for (int li = 0; li < 4; ++li) {
        const int l = threadIdx.x + li * 256;
        float s = 0.f;
#pragma unroll
        for (int a = 0; a < A_; ++a) s += ha[(size_t)a * L_ + l] * ta[(size_t)a * L_ + l];
        s *= (1.f / A_);
        loc[li] = s;
        lsum += s;
    }
    __shared__ float red[256];
    red[threadIdx.x] = lsum;
    __syncthreads();
    for (int off = 128; off > 0; off >>= 1) {
        if (threadIdx.x < off) red[threadIdx.x] += red[threadIdx.x + off];
        __syncthreads();
    }
    const float inv = 1.f / (red[0] + 1e-30f);
#pragma unroll
    for (int li = 0; li < 4; ++li)
        ht_bf[(size_t)bp * L_ + threadIdx.x + li * 256] = (__bf16)(loc[li] * inv);
}

__global__ void build_cat1(const float* __restrict__ ent_emb,
                           const __bf16* __restrict__ rs,
                           const int* __restrict__ hts,
                           __bf16* __restrict__ hs_cat,
                           __bf16* __restrict__ ts_cat)
{
    const int n = blockIdx.x;
    const int b = n / P_;
    const int hI = hts[n * 2 + 0];
    const int tI = hts[n * 2 + 1];
    const int j = threadIdx.x;                      // 768 threads
    const __bf16 rv = rs[(size_t)n * H_ + j];
    hs_cat[(size_t)n * 1536 + j]       = (__bf16)ent_emb[(size_t)(b * E_ + hI) * H_ + j];
    hs_cat[(size_t)n * 1536 + H_ + j]  = rv;
    ts_cat[(size_t)n * 1536 + j]       = (__bf16)ent_emb[(size_t)(b * E_ + tI) * H_ + j];
    ts_cat[(size_t)n * 1536 + H_ + j]  = rv;
}

__global__ void build_cat2(const __bf16* __restrict__ hs1, const __bf16* __restrict__ ts1,
                           const __bf16* __restrict__ rs,  const __bf16* __restrict__ feat,
                           __bf16* __restrict__ ch, __bf16* __restrict__ ct)
{
    const int n = blockIdx.x;
    const int j = threadIdx.x;                     // 768 threads
    const __bf16 rv = rs[(size_t)n * H_ + j];
    const __bf16 fv = feat[(size_t)n * H_ + j];
    ch[(size_t)n * 2304 + j]            = hs1[(size_t)n * H_ + j];
    ch[(size_t)n * 2304 + H_ + j]       = rv;
    ch[(size_t)n * 2304 + 2 * H_ + j]   = fv;
    ct[(size_t)n * 2304 + j]            = ts1[(size_t)n * H_ + j];
    ct[(size_t)n * 2304 + H_ + j]       = rv;
    ct[(size_t)n * 2304 + 2 * H_ + j]   = fv;
}

// =====================================================================
// Launcher
// =====================================================================
extern "C" void kernel_launch(void* const* d_in, const int* in_sizes, int n_in,
                              void* d_out, int out_size, void* d_ws, size_t ws_size,
                              hipStream_t stream)
{
    const float* seq  = (const float*)d_in[0];
    const float* att  = (const float*)d_in[1];
    const float* mask = (const float*)d_in[2];
    const float* Wh   = (const float*)d_in[3];
    const float* bh   = (const float*)d_in[4];
    const float* Wt   = (const float*)d_in[5];
    const float* bt   = (const float*)d_in[6];
    const float* Wh2  = (const float*)d_in[7];
    const float* bh2  = (const float*)d_in[8];
    const float* Wt2  = (const float*)d_in[9];
    const float* bt2  = (const float*)d_in[10];
    const float* W2   = (const float*)d_in[11];
    const float* b2   = (const float*)d_in[12];
    const float* Wb   = (const float*)d_in[13];
    const float* bb   = (const float*)d_in[14];
    const int*   midx = (const int*)d_in[15];
    const int*   hts  = (const int*)d_in[16];
    float* out = (float*)d_out;

    // ---- workspace carve-out ----
    size_t off = 0;
    char* wsb = (char*)d_ws;
    auto carve = [&](size_t bytes) -> void* {
        void* p = wsb + off;
        off = (off + bytes + 255) & ~(size_t)255;
        return p;
    };
    float*  ent_emb = (float*)carve((size_t)B_ * E_ * H_ * 4);
    float*  ent_att = (float*)carve((size_t)B_ * E_ * A_ * L_ * 4);
    __bf16* seqT    = (__bf16*)carve((size_t)B_ * H_ * L_ * 2);   // (b, H, L)
    __bf16* ht_bf   = (__bf16*)carve((size_t)NP_ * L_ * 2);
    __bf16* rs_bf   = (__bf16*)carve((size_t)NP_ * H_ * 2);
    __bf16* WhT     = (__bf16*)carve((size_t)EMBD * 1536 * 2);    // (N,K)
    __bf16* WtT     = (__bf16*)carve((size_t)EMBD * 1536 * 2);
    __bf16* Wh2T    = (__bf16*)carve((size_t)EMBD * 2304 * 2);
    __bf16* Wt2T    = (__bf16*)carve((size_t)EMBD * 2304 * 2);
    __bf16* W2T     = (__bf16*)carve((size_t)H_ * KB_ * 2);
    __bf16* WbT     = (__bf16*)carve((size_t)NLP_ * KB_ * 2);     // padded rows
    __bf16* hs_cat  = (__bf16*)carve((size_t)NP_ * 1536 * 2);
    __bf16* ts_cat  = (__bf16*)carve((size_t)NP_ * 1536 * 2);
    __bf16* hs1     = (__bf16*)carve((size_t)NP_ * EMBD * 2);
    __bf16* ts1     = (__bf16*)carve((size_t)NP_ * EMBD * 2);
    __bf16* feat    = (__bf16*)carve((size_t)NP_ * H_ * 2);
    __bf16* ch2     = (__bf16*)carve((size_t)NP_ * 2304 * 2);
    __bf16* ct2     = (__bf16*)carve((size_t)NP_ * 2304 * 2);
    __bf16* hs2     = (__bf16*)carve((size_t)NP_ * EMBD * 2);
    __bf16* ts2     = (__bf16*)carve((size_t)NP_ * EMBD * 2);

    auto cvtT = [&](const float* src, __bf16* dst, int K, int N, int Nout, int batch) {
        dim3 g(Nout / 32, K / 32, batch);
        cvt_transpose_bf16<<<g, 256, 0, stream>>>(src, dst, K, N, Nout);
    };

    // ---- stage 0: transpose-convert all GEMM B operands to bf16 (N,K) ----
    cvtT(seq, seqT, L_, H_, H_, B_);          // per-batch seq^T (768,1024)
    cvtT(Wh,  WhT,  1536, EMBD, EMBD, 1);
    cvtT(Wt,  WtT,  1536, EMBD, EMBD, 1);
    cvtT(Wh2, Wh2T, 2304, EMBD, EMBD, 1);
    cvtT(Wt2, Wt2T, 2304, EMBD, EMBD, 1);
    cvtT(W2,  W2T,  KB_,  H_,   H_,   1);
    cvtT(Wb,  WbT,  KB_,  NL_,  NLP_, 1);     // zero-padded to 128 rows

    // ---- stage 1: entity embeddings / attentions ----
    ent_emb_kernel<<<B_ * E_, 256, 0, stream>>>(seq, mask, midx, ent_emb);
    ent_att_kernel<<<B_ * E_ * A_, 256, 0, stream>>>(att, mask, midx, ent_att);
    ht_att_kernel<<<NP_, 256, 0, stream>>>(ent_att, hts, ht_bf);

    // ---- stage 2: rs = ht_att @ seq  (per batch) ----
    for (int b = 0; b < B_; ++b) {
        dim3 grid(H_ / 128, (P_ + 127) / 128);
        gemm_bf16_wmma<<<grid, 256, 0, stream>>>(
            ht_bf + (size_t)b * P_ * L_, seqT + (size_t)b * H_ * L_, nullptr,
            rs_bf + (size_t)b * P_ * H_, nullptr,
            P_, H_, L_, H_, H_, 0);
    }

    // ---- stage 3: hs1/ts1 = tanh([ent_emb|rs] @ Wh/Wt + b) ----
    build_cat1<<<NP_, H_, 0, stream>>>(ent_emb, rs_bf, hts, hs_cat, ts_cat);
    {
        dim3 grid(EMBD / 128, (NP_ + 127) / 128);
        gemm_bf16_wmma<<<grid, 256, 0, stream>>>(hs_cat, WhT, bh, hs1, nullptr,
                                                 NP_, EMBD, 1536, EMBD, EMBD, 1);
        gemm_bf16_wmma<<<grid, 256, 0, stream>>>(ts_cat, WtT, bt, ts1, nullptr,
                                                 NP_, EMBD, 1536, EMBD, EMBD, 1);
    }

    // ---- stage 4: hts_feat = blockbilinear(hs1,ts1) @ W2 + b2  (fused, no X) ----
    {
        dim3 grid(H_ / 128, (NP_ + 127) / 128);
        gemm_bilinear_wmma<<<grid, 256, 0, stream>>>(hs1, ts1, W2T, b2, feat, nullptr,
                                                     NP_, H_, H_);
    }

    // ---- stage 5: hs2/ts2 = tanh([x1|rs|feat] @ Wh2/Wt2 + b) ----
    build_cat2<<<NP_, H_, 0, stream>>>(hs1, ts1, rs_bf, feat, ch2, ct2);
    {
        dim3 grid(EMBD / 128, (NP_ + 127) / 128);
        gemm_bf16_wmma<<<grid, 256, 0, stream>>>(ch2, Wh2T, bh2, hs2, nullptr,
                                                 NP_, EMBD, 2304, EMBD, EMBD, 1);
        gemm_bf16_wmma<<<grid, 256, 0, stream>>>(ct2, Wt2T, bt2, ts2, nullptr,
                                                 NP_, EMBD, 2304, EMBD, EMBD, 1);
    }

    // ---- stage 6: logits = blockbilinear(hs2,ts2) @ Wb + bb  (fused, f32 out) ----
    {
        dim3 grid(NLP_ / 128, (NP_ + 127) / 128);
        gemm_bilinear_wmma<<<grid, 256, 0, stream>>>(hs2, ts2, WbT, bb, nullptr, out,
                                                     NP_, NL_, NL_);
    }
}
